// OverlappedMoE_42649025249863
// MI455X (gfx1250) — matrix operand order repaired and checked by reading, not compiled, not run on hardware
//
#include <hip/hip_runtime.h>

#define H       1024
#define EXPERTS 8
#define NTOK    16384   // B*S = 4*4096

#define BM 64
#define BN 256
#define BK 32
#define AST 40   // LDS row stride (halfs), padded vs 32 to spread banks
#define BST 40

typedef __attribute__((ext_vector_type(16))) __bf16       v16bf;
typedef __attribute__((ext_vector_type(8)))  float        v8f;
typedef __attribute__((ext_vector_type(4)))  unsigned int v4u;

union FragU { v16bf v; v4u q[2]; };
union H8    { v4u q; unsigned short h[8]; };

__device__ __forceinline__ unsigned short f2bf(float f) {
  unsigned int u = __builtin_bit_cast(unsigned int, f);
  u += 0x7FFFu + ((u >> 16) & 1u);            // round-to-nearest-even
  return (unsigned short)(u >> 16);
}
__device__ __forceinline__ float bf2f(unsigned short s) {
  unsigned int u = ((unsigned int)s) << 16;
  return __builtin_bit_cast(float, u);
}

// 32-bit LDS byte offset from a generic pointer to a __shared__ object
// (shared aperture: addr[31:0] == LDS offset)
__device__ __forceinline__ unsigned lds_off(const void* p) {
  return (unsigned)(size_t)p;
}
// async 16B global -> LDS copy, tracked with ASYNCcnt (CDNA5).
// volatile (ordered vs other volatile asm: the s_wait below) but NO memory
// clobber, so the scheduler can still unroll/pipeline around it.
__device__ __forceinline__ void async_b128(unsigned lds, const void* gaddr) {
  asm volatile("global_load_async_to_lds_b128 %0, %1, off"
               :: "v"(lds), "v"(gaddr));
}
__device__ __forceinline__ void wait_async0() {
  asm volatile("s_wait_asynccnt 0" ::: "memory");
}

// ---------------------------------------------------------------- utilities
__global__ void moe_zero_counts(int* counts) {
  if (threadIdx.x < EXPERTS) counts[threadIdx.x] = 0;
}

__global__ void moe_cvt_bf16(const float* __restrict__ src,
                             unsigned short* __restrict__ dst, int n) {
  for (int i = blockIdx.x * blockDim.x + threadIdx.x; i < n;
       i += gridDim.x * blockDim.x)
    dst[i] = f2bf(src[i]);
}

// ------------------------------------------------------------------- gating
// one wave (32 lanes) per token; block = 256 threads = 8 tokens
__global__ __launch_bounds__(256)
void moe_gate(const float* __restrict__ tokens, const float* __restrict__ gate_w,
              int* __restrict__ counts, int* __restrict__ lists,
              float* __restrict__ gw2) {
  __shared__ float s_gw[EXPERTS * H];   // 32 KB
  const int tid = threadIdx.x;
  for (int i = tid; i < EXPERTS * H; i += 256) s_gw[i] = gate_w[i];
  __syncthreads();

  const int lane = tid & 31;
  const int wave = tid >> 5;
  const int n = blockIdx.x * 8 + wave;
  const float* x = tokens + (size_t)n * H;

  float acc[EXPERTS];
#pragma unroll
  for (int e = 0; e < EXPERTS; ++e) acc[e] = 0.0f;

  for (int h = lane; h < H; h += 32) {
    float xv = x[h];
#pragma unroll
    for (int e = 0; e < EXPERTS; ++e) acc[e] += xv * s_gw[e * H + h];
  }
#pragma unroll
  for (int e = 0; e < EXPERTS; ++e)
#pragma unroll
    for (int off = 16; off > 0; off >>= 1)
      acc[e] += __shfl_xor(acc[e], off, 32);

  if (lane == 0) {
    int e0 = 0; float m0 = acc[0];
#pragma unroll
    for (int e = 1; e < EXPERTS; ++e)
      if (acc[e] > m0) { m0 = acc[e]; e0 = e; }
    float sum = 0.0f;
#pragma unroll
    for (int e = 0; e < EXPERTS; ++e) sum += __expf(acc[e] - m0);
    int e1 = -1; float m1 = -3.4e38f;
#pragma unroll
    for (int e = 0; e < EXPERTS; ++e)
      if (e != e0 && acc[e] > m1) { m1 = acc[e]; e1 = e; }
    float w0 = 1.0f / sum;               // exp(0)/sum
    float w1 = __expf(m1 - m0) / sum;
    int p0 = atomicAdd(&counts[e0], 1);
    lists[e0 * NTOK + p0] = 2 * n;       // slot = token*2 + rank
    int p1 = atomicAdd(&counts[e1], 1);
    lists[e1 * NTOK + p1] = 2 * n + 1;
    gw2[2 * n]     = w0;
    gw2[2 * n + 1] = w1;
  }
}

// -------------------------------------------------- sparse per-expert GEMM
// block: 256 thr = 8 waves (4 along M x 2 along N); tile BM=64 x BN=256
// async global->LDS, explicit 2x-unrolled ping-pong (constant LDS offsets)
__global__ __launch_bounds__(256)
void moe_expert_gemm(const unsigned short* __restrict__ tokbf,
                     const unsigned short* __restrict__ expw,
                     const float* __restrict__ expert_b,
                     const int* __restrict__ lists,
                     const int* __restrict__ counts,
                     unsigned short* __restrict__ y2) {
  const int e     = blockIdx.z;
  const int cnt   = counts[e];
  const int mBase = blockIdx.y * BM;
  if (mBase >= cnt) return;                       // empty tile -> exit
  const int nBase = blockIdx.x * BN;

  __shared__ int            s_tok[BM];
  __shared__ float          s_bias[BN];
  __shared__ unsigned short sA[2][BM * AST];
  __shared__ unsigned short sB[2][BN * BST];

  const int tid  = threadIdx.x;
  const int lane = tid & 31, wave = tid >> 5;
  const int wm   = wave >> 1, wn = wave & 1;

  if (tid < BM) {
    int r = mBase + tid;
    s_tok[tid] = lists[e * NTOK + (r < cnt ? r : cnt - 1)];
  }
  s_bias[tid] = expert_b[e * H + nBase + tid];
  __syncthreads();                                 // s_tok visible to stagers

  const int lrow = tid >> 2;        // 0..63
  const int kof  = (tid & 3) * 8;   // 0,8,16,24 halfs
  const unsigned short* wbase = expw + (size_t)e * H * H;
  const unsigned short* arow  = tokbf + (size_t)(s_tok[lrow] >> 1) * H + kof;

  v8f acc[8];
#pragma unroll
  for (int i = 0; i < 8; ++i)
#pragma unroll
    for (int j = 0; j < 8; ++j) acc[i][j] = 0.0f;

  const int am = wm * 16 + (lane & 15);
  const int ak = (lane >> 4) * 8;
  const int bk = (lane >> 4) * 8;
  const int bn0 = wn * 128 + (lane & 15);

  auto stage = [&](int kk, int buf) {
    async_b128(lds_off(&sA[buf][lrow * AST + kof]), arow + kk);
#pragma unroll
    for (int p = 0; p < 4; ++p) {
      int row = lrow + p * 64;
      async_b128(lds_off(&sB[buf][row * BST + kof]),
                 wbase + (size_t)(nBase + row) * H + kk + kof);
    }
  };
  auto compute = [&](int buf) {
    FragU a;
    a.q[0] = *reinterpret_cast<const v4u*>(&sA[buf][am * AST + ak]);
    a.q[1] = *reinterpret_cast<const v4u*>(&sA[buf][am * AST + ak + 16]);
    FragU b[8];
#pragma unroll
    for (int i = 0; i < 8; ++i) {
      const int bn = bn0 + i * 16;
      b[i].q[0] = *reinterpret_cast<const v4u*>(&sB[buf][bn * BST + bk]);
      b[i].q[1] = *reinterpret_cast<const v4u*>(&sB[buf][bn * BST + bk + 16]);
    }
#pragma unroll
    for (int i = 0; i < 8; ++i)
      acc[i] = __builtin_amdgcn_wmma_f32_16x16x32_bf16(
          false, a.v, false, b[i].v, (short)0, acc[i], false, false);
  };

  stage(0, 0);
  wait_async0();
  __syncthreads();

  for (int kk = 0; kk < H; kk += 2 * BK) {        // ping-pong, literal bufs
    if (kk + BK < H) stage(kk + BK, 1);
    compute(0);
    wait_async0();
    __syncthreads();
    if (kk + 2 * BK < H) stage(kk + 2 * BK, 0);
    compute(1);
    wait_async0();
    __syncthreads();
  }

  // epilogue: bias + SiLU -> bf16 scatter into slot rows
  const int rowoff = (lane < 16) ? 0 : 8;
  const int coll   = lane & 15;
#pragma unroll
  for (int i = 0; i < 8; ++i) {
    const int col  = nBase + wn * 128 + i * 16 + coll;
    const float bias = s_bias[col - nBase];
#pragma unroll
    for (int j = 0; j < 8; ++j) {
      const int rloc = wm * 16 + rowoff + j;
      if (mBase + rloc < cnt) {
        const int slot = s_tok[rloc];
        float v = acc[i][j] + bias;
        v = v / (1.0f + __expf(-v));          // SiLU
        y2[(size_t)slot * H + col] = f2bf(v);
      }
    }
  }
}

// ------------------------------------------------------------ combine GEMM
// out[n,d] = sum_h (w0*y2[2n,h] + w1*y2[2n+1,h]) * combine_w[d,h]
__global__ __launch_bounds__(256)
void moe_combine_gemm(const unsigned short* __restrict__ y2,
                      const unsigned short* __restrict__ combw,
                      const float* __restrict__ gw2,
                      float* __restrict__ out) {
  const int mBase = blockIdx.y * BM;
  const int nBase = blockIdx.x * BN;

  __shared__ float          s_w[BM][2];
  __shared__ unsigned short sA[2][BM * AST];
  __shared__ unsigned short sB[2][BN * BST];

  const int tid  = threadIdx.x;
  const int lane = tid & 31, wave = tid >> 5;
  const int wm   = wave >> 1, wn = wave & 1;

  if (tid < BM * 2)
    s_w[tid >> 1][tid & 1] = gw2[(mBase + (tid >> 1)) * 2 + (tid & 1)];
  __syncthreads();                               // s_w visible to stagers

  const int lrow = tid >> 2;
  const int kof  = (tid & 3) * 8;
  const float w0 = s_w[lrow][0], w1 = s_w[lrow][1];
  const unsigned short* y2r0 = y2 + (size_t)(2 * (mBase + lrow)) * H + kof;
  const unsigned short* y2r1 = y2r0 + H;

  v8f acc[8];
#pragma unroll
  for (int i = 0; i < 8; ++i)
#pragma unroll
    for (int j = 0; j < 8; ++j) acc[i][j] = 0.0f;

  const int am = wm * 16 + (lane & 15);
  const int ak = (lane >> 4) * 8;
  const int bk = (lane >> 4) * 8;
  const int bn0 = wn * 128 + (lane & 15);

  auto stageB = [&](int kk, int buf) {           // async path
#pragma unroll
    for (int p = 0; p < 4; ++p) {
      int row = lrow + p * 64;
      async_b128(lds_off(&sB[buf][row * BST + kof]),
                 combw + (size_t)(nBase + row) * H + kk + kof);
    }
  };
  auto stageA = [&](int kk, int buf) {           // VALU mix path
    H8 q0, q1, o;
    q0.q = *reinterpret_cast<const v4u*>(y2r0 + kk);
    q1.q = *reinterpret_cast<const v4u*>(y2r1 + kk);
#pragma unroll
    for (int t = 0; t < 8; ++t)
      o.h[t] = f2bf(w0 * bf2f(q0.h[t]) + w1 * bf2f(q1.h[t]));
    *reinterpret_cast<v4u*>(&sA[buf][lrow * AST + kof]) = o.q;
  };
  auto compute = [&](int buf) {
    FragU a;
    a.q[0] = *reinterpret_cast<const v4u*>(&sA[buf][am * AST + ak]);
    a.q[1] = *reinterpret_cast<const v4u*>(&sA[buf][am * AST + ak + 16]);
    FragU b[8];
#pragma unroll
    for (int i = 0; i < 8; ++i) {
      const int bn = bn0 + i * 16;
      b[i].q[0] = *reinterpret_cast<const v4u*>(&sB[buf][bn * BST + bk]);
      b[i].q[1] = *reinterpret_cast<const v4u*>(&sB[buf][bn * BST + bk + 16]);
    }
#pragma unroll
    for (int i = 0; i < 8; ++i)
      acc[i] = __builtin_amdgcn_wmma_f32_16x16x32_bf16(
          false, a.v, false, b[i].v, (short)0, acc[i], false, false);
  };

  stageB(0, 0);
  stageA(0, 0);
  wait_async0();
  __syncthreads();

  for (int kk = 0; kk < H; kk += 2 * BK) {
    if (kk + BK < H) { stageB(kk + BK, 1); stageA(kk + BK, 1); }
    compute(0);
    wait_async0();
    __syncthreads();
    if (kk + 2 * BK < H) { stageB(kk + 2 * BK, 0); stageA(kk + 2 * BK, 0); }
    compute(1);
    wait_async0();
    __syncthreads();
  }

  const int rowoff = (lane < 16) ? 0 : 8;
  const int coll   = lane & 15;
#pragma unroll
  for (int i = 0; i < 8; ++i) {
    const int col = nBase + wn * 128 + i * 16 + coll;
#pragma unroll
    for (int j = 0; j < 8; ++j) {
      const int n = mBase + wm * 16 + rowoff + j;
      out[(size_t)n * H + col] = acc[i][j];
    }
  }
}

// ---------------------------------------------------------------- launcher
extern "C" void kernel_launch(void* const* d_in, const int* in_sizes, int n_in,
                              void* d_out, int out_size, void* d_ws, size_t ws_size,
                              hipStream_t stream) {
  const float* tokens    = (const float*)d_in[0];
  const float* gate_w    = (const float*)d_in[1];
  const float* expert_w  = (const float*)d_in[2];
  const float* expert_b  = (const float*)d_in[3];
  const float* combine_w = (const float*)d_in[4];
  float* out = (float*)d_out;

  char* ws = (char*)d_ws;
  size_t off = 0;
  int* counts = (int*)(ws + off);                 off += 256;
  int* lists  = (int*)(ws + off);                 off += (size_t)EXPERTS * NTOK * 4;
  float* gw2  = (float*)(ws + off);               off += (size_t)NTOK * 2 * 4;
  unsigned short* tokbf  = (unsigned short*)(ws + off); off += (size_t)NTOK * H * 2;
  unsigned short* expwbf = (unsigned short*)(ws + off); off += (size_t)EXPERTS * H * H * 2;
  unsigned short* combwbf= (unsigned short*)(ws + off); off += (size_t)H * H * 2;
  unsigned short* y2     = (unsigned short*)(ws + off); off += (size_t)NTOK * 2 * H * 2;

  moe_zero_counts<<<1, 64, 0, stream>>>(counts);
  moe_cvt_bf16<<<2048, 256, 0, stream>>>(tokens,    tokbf,   NTOK * H);
  moe_cvt_bf16<<<2048, 256, 0, stream>>>(expert_w,  expwbf,  EXPERTS * H * H);
  moe_cvt_bf16<<<512,  256, 0, stream>>>(combine_w, combwbf, H * H);
  moe_gate<<<NTOK / 8, 256, 0, stream>>>(tokens, gate_w, counts, lists, gw2);
  moe_expert_gemm<<<dim3(H / BN, NTOK / BM, EXPERTS), 256, 0, stream>>>(
      tokbf, expwbf, expert_b, lists, counts, y2);
  moe_combine_gemm<<<dim3(H / BN, NTOK / BM), 256, 0, stream>>>(
      y2, combwbf, gw2, out);
}
